// ExclusiveConv_34857954574524
// MI455X (gfx1250) — matrix operand order repaired and checked by reading, not compiled, not run on hardware
//
#include <hip/hip_runtime.h>

#define C_IN   64
#define C_OUT  64
#define KTAPS  9
#define KFLAT  (KTAPS * C_IN)        // 576 = flattened K dimension (tap*64 + c)
#define NCHUNK (KFLAT / 32)          // 18 chunks of K=32
#define CH_PER_PASS 9                // two LDS passes of 9 chunks -> 36 KB LDS each
#define NT     4                     // 4 column tiles of 16 -> 64 output channels

typedef __attribute__((ext_vector_type(16))) _Float16 v16h;
typedef __attribute__((ext_vector_type(8)))  float    v8f;

// ---------------------------------------------------------------------------
// Kernel 0: rate[c] = softplus(decay_rate[c])   (64 values)
// ---------------------------------------------------------------------------
__global__ void ExclusiveConv_rate_kernel(const float* __restrict__ dr,
                                          float* __restrict__ rate) {
    int c = threadIdx.x;
    if (c < C_IN) {
        float x = dr[c];
        rate[c] = (x > 20.0f) ? x : log1pf(__expf(x));
    }
}

// ---------------------------------------------------------------------------
// Kernel 1: scatter.  One thread per (event, 4-channel group).
// nd layout: interleaved pairs  nd[2*(seg*64+c)] = num, nd[2*(seg*64+c)+1] = den
// -> each thread's 8 f32 atomics span 32 contiguous bytes.
// ---------------------------------------------------------------------------
__global__ void ExclusiveConv_scatter_kernel(const float* __restrict__ feat,
                                             const float* __restrict__ dt,
                                             const int*   __restrict__ skid,
                                             const int*   __restrict__ segid,
                                             const float* __restrict__ rate,
                                             float*       __restrict__ nd,
                                             int e_in) {
    int t = blockIdx.x * blockDim.x + threadIdx.x;
    int e = t >> 4;
    if (e >= e_in) return;
    int c = (t & 15) * 4;

    float dte = dt[e];
    long long s = (long long)segid[e] * KTAPS + (long long)skid[e];

    const float4 f = *(const float4*)(feat + (long long)e * C_IN + c);
    const float4 r = *(const float4*)(rate + c);

    float w0 = __expf(-dte * r.x);
    float w1 = __expf(-dte * r.y);
    float w2 = __expf(-dte * r.z);
    float w3 = __expf(-dte * r.w);

    float* base = nd + (s * C_IN + c) * 2;
    __hip_atomic_fetch_add(base + 0, w0 * f.x, __ATOMIC_RELAXED, __HIP_MEMORY_SCOPE_AGENT);
    __hip_atomic_fetch_add(base + 1, w0,       __ATOMIC_RELAXED, __HIP_MEMORY_SCOPE_AGENT);
    __hip_atomic_fetch_add(base + 2, w1 * f.y, __ATOMIC_RELAXED, __HIP_MEMORY_SCOPE_AGENT);
    __hip_atomic_fetch_add(base + 3, w1,       __ATOMIC_RELAXED, __HIP_MEMORY_SCOPE_AGENT);
    __hip_atomic_fetch_add(base + 4, w2 * f.z, __ATOMIC_RELAXED, __HIP_MEMORY_SCOPE_AGENT);
    __hip_atomic_fetch_add(base + 5, w2,       __ATOMIC_RELAXED, __HIP_MEMORY_SCOPE_AGENT);
    __hip_atomic_fetch_add(base + 6, w3 * f.w, __ATOMIC_RELAXED, __HIP_MEMORY_SCOPE_AGENT);
    __hip_atomic_fetch_add(base + 7, w3,       __ATOMIC_RELAXED, __HIP_MEMORY_SCOPE_AGENT);
}

// ---------------------------------------------------------------------------
// Kernel 2: normalize + GEMM via v_wmma_f32_16x16x32_f16.
// One wave owns a 16-row x 64-col output tile (4 accumulators of 16 cols).
// A fragment (16x32 f16): lane L (rowA=L&15, half=L>>4), VGPR v holds
// K = kk + (v>>2)*16 + half*8 + (v&3)*2 -> one aligned float4 load gives
// {num,den} for K and K+1.  Normalization is branchless v_rcp + v_cndmask.
// B (576x64 weights) pre-swizzled in LDS in exact B-fragment layout
// (2 passes x 36 KB), read back as ds_load_b128 pairs.
// A loads are software-pipelined one chunk ahead of the WMMAs.
// ---------------------------------------------------------------------------
__device__ __forceinline__ void load_q8(float4* q, const float* aggrow,
                                        int kk, int half) {
    #pragma unroll
    for (int v = 0; v < 8; ++v) {
        const int kidx = kk + (v >> 2) * 16 + half * 8 + (v & 3) * 2;
        q[v] = *(const float4*)(aggrow + kidx * 2);
    }
}

__device__ __forceinline__ v16h convert_a(const float4* q) {
    v16h a;
    #pragma unroll
    for (int v = 0; v < 8; ++v) {
        const float4 p = q[v];
        const float inv_lo = __builtin_amdgcn_rcpf(p.y);
        const float inv_hi = __builtin_amdgcn_rcpf(p.w);
        const float a_lo = (p.y > 0.0f) ? p.x * inv_lo : 0.0f;  // v_cndmask
        const float a_hi = (p.w > 0.0f) ? p.z * inv_hi : 0.0f;
        a[2 * v]     = (_Float16)a_lo;
        a[2 * v + 1] = (_Float16)a_hi;
    }
    return a;
}

__global__ void __launch_bounds__(256)
ExclusiveConv_gemm_kernel(const float* __restrict__ nd,
                          const float* __restrict__ kern,   // [9][64][64] f32
                          const float* __restrict__ bias,
                          float*       __restrict__ out,
                          int e_out) {
    __shared__ __align__(16) _Float16 bsh[CH_PER_PASS * NT * 32 * 16]; // 36 KB

    const int lane = threadIdx.x & 31;
    const int wave = threadIdx.x >> 5;
    const int tile = blockIdx.x * 8 + wave;
    const int rowbase = tile * 16;
    const bool active = (rowbase < e_out);      // wave-uniform

    const int rowA = lane & 15;
    const int half = lane >> 4;
    int rowR = rowbase + rowA;
    if (rowR >= e_out) rowR = e_out - 1;        // clamp (loads only)
    const float* aggrow = nd + (long long)rowR * (KFLAT * 2);

    v8f acc[NT] = {{}, {}, {}, {}};

    for (int pass = 0; pass < 2; ++pass) {
        // ---- fill swizzled B fragments for 9 K-chunks (all threads) ----
        for (int s = threadIdx.x; s < CH_PER_PASS * NT * 32; s += 256) {
            int chl = s / (NT * 32);
            int rem = s - chl * (NT * 32);
            int nt  = rem >> 5;
            int L   = rem & 31;
            int col   = nt * 16 + (L & 15);
            int kbase = (pass * CH_PER_PASS + chl) * 32 + (L >> 4) * 8;
            _Float16* dst = &bsh[s * 16];
            #pragma unroll
            for (int tt = 0; tt < 16; ++tt)
                dst[tt] = (_Float16)kern[(long long)(kbase + tt + (tt >= 8 ? 8 : 0)) * C_OUT + col];
        }
        __syncthreads();

        if (active) {
            float4 q[8];
            load_q8(q, aggrow, pass * CH_PER_PASS * 32, half);   // chunk 0 of pass
            for (int chl = 0; chl < CH_PER_PASS; ++chl) {
                const v16h a = convert_a(q);
                if (chl + 1 < CH_PER_PASS)                        // prefetch next chunk
                    load_q8(q, aggrow, (pass * CH_PER_PASS + chl + 1) * 32, half);
                #pragma unroll
                for (int nt = 0; nt < NT; ++nt) {
                    const v16h b = *(const v16h*)(&bsh[((chl * NT + nt) * 32 + lane) * 16]);
                    acc[nt] = __builtin_amdgcn_wmma_f32_16x16x32_f16(
                        false, a, false, b, (short)0, acc[nt], false, false);
                }
            }
        }
        __syncthreads();
    }

    if (active) {
        #pragma unroll
        for (int nt = 0; nt < NT; ++nt) {
            const int col = nt * 16 + (lane & 15);
            const float bv = bias[col];
            #pragma unroll
            for (int r = 0; r < 8; ++r) {
                const int row = rowbase + r + 8 * half;   // C/D layout: M = r + 8*half
                if (row < e_out)
                    out[(long long)row * C_OUT + col] = acc[nt][r] + bv;
            }
        }
    }
}

// ---------------------------------------------------------------------------
extern "C" void kernel_launch(void* const* d_in, const int* in_sizes, int n_in,
                              void* d_out, int out_size, void* d_ws, size_t ws_size,
                              hipStream_t stream) {
    const float* features   = (const float*)d_in[0];
    const float* dt         = (const float*)d_in[1];
    /* d_in[2] = times_out: only its length matters */
    const float* decay_rate = (const float*)d_in[3];
    const float* kernel_w   = (const float*)d_in[4];
    const float* bias       = (const float*)d_in[5];
    const int*   skid       = (const int*)d_in[6];
    const int*   segid      = (const int*)d_in[7];
    float*       out        = (float*)d_out;

    const int e_in  = in_sizes[0] / C_IN;
    const int e_out = in_sizes[2];

    // workspace: interleaved {num,den} pairs, then rate[64]
    const size_t nd_elems = (size_t)e_out * KTAPS * C_IN * 2;
    float* nd   = (float*)d_ws;
    float* rate = nd + nd_elems;

    // 0) rate = softplus(decay_rate)
    ExclusiveConv_rate_kernel<<<1, 64, 0, stream>>>(decay_rate, rate);

    // 1) zero accumulators (graph-capturable async memset)
    hipMemsetAsync(nd, 0, nd_elems * sizeof(float), stream);

    // 2) scatter with f32 global atomics
    {
        long long threads = (long long)e_in * 16;
        int blocks = (int)((threads + 255) / 256);
        ExclusiveConv_scatter_kernel<<<blocks, 256, 0, stream>>>(
            features, dt, skid, segid, rate, nd, e_in);
    }

    // 3) fused normalize + WMMA GEMM + bias
    {
        int tiles = (e_out + 15) / 16;
        int wgs   = (tiles + 7) / 8;
        ExclusiveConv_gemm_kernel<<<wgs, 256, 0, stream>>>(
            nd, kernel_w, bias, out, e_out);
    }
}
